// FlowSiteModel_31001073943180
// MI455X (gfx1250) — compile-verified, matrix operand order checked by reading, not run on hardware
//
#include <hip/hip_runtime.h>

typedef __attribute__((ext_vector_type(16))) _Float16 v16h;
typedef __attribute__((ext_vector_type(8)))  _Float16 v8h;
typedef __attribute__((ext_vector_type(8)))  float    v8f;

#define RBF   64
#define NCH   5
#define K1    320            // 5*64
#define FD    128
#define MT    64             // edges per block (four 16-row WMMA sub-tiles)
#define NSUB  4              // MT/16
#define ASTR  328            // attr row stride in halves (320 + 8 pad)
#define HSTR  136            // h row stride in halves (128 + 8 pad)
#define KC1   10             // 320/32 k-chunks for GEMM1
#define KC2   4              // 128/32 k-chunks for GEMM2
#define PK1   (KC1*8*32*16)  // 40960 halves
#define PK2   (KC2*8*32*16)  // 16384 halves

// ---- fragment loaders -------------------------------------------------------
// 16-bit A-matrix 16x32 layout (ISA 7.12.2):
//   lanes 0-15 : M=lane,    halves 0-7 -> K 0..7,  halves 8-15 -> K 16..23
//   lanes 16-31: M=lane-16, halves 0-7 -> K 8..15, halves 8-15 -> K 24..31
// Row-major f16 tile => two contiguous 16B LDS loads per lane.
__device__ __forceinline__ v16h load_a_frag(const _Float16* base, int stride,
                                            int lane, int kbase) {
  const int row = lane & 15;
  const int hi  = (lane >> 4) & 1;
  const _Float16* p = base + row * stride + kbase + hi * 8;
  union { v16h v; v8h h[2]; } u;
  u.h[0] = *(const v8h*)(p);        // K .. K+7
  u.h[1] = *(const v8h*)(p + 16);   // K+16 .. K+23
  return u.v;
}

// B fragments are pre-packed contiguously per lane (16 halves = 32 bytes).
__device__ __forceinline__ v16h load_b_frag(const _Float16* p) {
  union { v16h v; v8h h[2]; } u;
  u.h[0] = *(const v8h*)(p);
  u.h[1] = *(const v8h*)(p + 8);
  return u.v;
}

// ---- weight pre-pack: f32 [K][N] -> per-lane f16 B fragments ----------------
// pack index t = ((kc*8 + ntile)*32 + lane)*16 + h
// B-matrix 32x16 layout (transpose-symmetric to A):
//   lanes 0-15 : N=lane,    h 0-7 -> K 0..7,  h 8-15 -> K 16..23
//   lanes 16-31: N=lane-16, h 0-7 -> K 8..15, h 8-15 -> K 24..31
__global__ __launch_bounds__(256) void pack_weights(
    const float* __restrict__ W1, const float* __restrict__ W2,
    _Float16* __restrict__ pack1, _Float16* __restrict__ pack2) {
  const int t = blockIdx.x * 256 + threadIdx.x;
  if (t < PK1) {
    const int h = t & 15, lane = (t >> 4) & 31, nt = (t >> 9) & 7, kc = t >> 12;
    const int K = kc * 32 + (h < 8 ? h : h + 8) + ((lane >> 4) << 3);
    const int N = nt * 16 + (lane & 15);
    pack1[t] = (_Float16)W1[K * FD + N];
  } else if (t < PK1 + PK2) {
    const int u = t - PK1;
    const int h = u & 15, lane = (u >> 4) & 31, nt = (u >> 9) & 7, kc = u >> 12;
    const int K = kc * 32 + (h < 8 ? h : h + 8) + ((lane >> 4) << 3);
    const int N = nt * 16 + (lane & 15);
    pack2[u] = (_Float16)W2[K * FD + N];
  }
}

// ---- fused edge kernel: gather -> dist -> RBF(f16) -> WMMA MLP --------------
__global__ __launch_bounds__(256) void fused_edge_mlp(
    const float* __restrict__ lig,
    const float* __restrict__ pCA, const float* __restrict__ pCb,
    const float* __restrict__ pC,  const float* __restrict__ pO,
    const float* __restrict__ pN,
    const float* __restrict__ b1,  const float* __restrict__ b2,
    const int*   __restrict__ src, const int* __restrict__ dst,
    const _Float16* __restrict__ pack1, const _Float16* __restrict__ pack2,
    float* __restrict__ out, int E)
{
  __shared__ __align__(16) _Float16 attr[MT * ASTR];   // 41984 B
  __shared__ __align__(16) _Float16 hbuf[MT * HSTR];   // 17408 B
  __shared__ float dists[MT * NCH];                    //  1280 B

  const int t    = threadIdx.x;
  const int tile = blockIdx.x * MT;

  // ---- Phase 1a: 5 distance channels for 64 edges (64 threads active) ----
  if (t < MT) {
    int ge = tile + t; if (ge >= E) ge = E - 1;
    const int si = src[ge] * 3, di = dst[ge] * 3;
    const float lx = lig[si], ly = lig[si + 1], lz = lig[si + 2];
    const float* P[NCH] = {pCA, pCb, pC, pO, pN};
#pragma unroll
    for (int ch = 0; ch < NCH; ++ch) {
      const float dx = lx - P[ch][di];
      const float dy = ly - P[ch][di + 1];
      const float dz = lz - P[ch][di + 2];
      dists[t * NCH + ch] = sqrtf(dx * dx + dy * dy + dz * dz);
    }
  }
  __syncthreads();

  // ---- Phase 1b: Gaussian smearing, 4 features per iteration, b64 stores ----
  const float DELTA = 30.0f / 63.0f;
  const float CO    = -0.5f / (DELTA * DELTA);
  for (int q = t; q < (MT * K1) / 4; q += 256) {
    const int idx = q * 4;                  // first feature index
    const int e   = idx / K1;
    const int f   = idx - e * K1;           // multiple of 4 -> same channel row
    const int ch  = f >> 6;
    const int j   = f & 63;
    const float d = dists[e * NCH + ch];
    union { _Float16 h[4]; unsigned long long u64; } pk;
#pragma unroll
    for (int s = 0; s < 4; ++s) {
      const float x = d - (float)(j + s) * DELTA;
      pk.h[s] = (_Float16)__expf(CO * x * x);
    }
    *(unsigned long long*)(&attr[e * ASTR + f]) = pk.u64;  // 8B-aligned
  }
  __syncthreads();

  const int w    = t >> 5;    // wave id = output column tile (8 * 16 = 128 cols)
  const int lane = t & 31;
  const int n    = w * 16 + (lane & 15);   // output column of this lane
  const int mh   = (lane >> 4) * 8;        // C/D layout: lanes 16-31 -> M+8

  // ---- GEMM1: attr[64x320] @ W1[320x128], f32 accumulate ----
  v8f acc[NSUB];
#pragma unroll
  for (int s = 0; s < NSUB; ++s) acc[s] = (v8f){};
#pragma unroll
  for (int kc = 0; kc < KC1; ++kc) {
    const v16h b = load_b_frag(pack1 + (((kc * 8 + w) * 32 + lane) << 4));
#pragma unroll
    for (int s = 0; s < NSUB; ++s) {
      const v16h a = load_a_frag(attr + s * 16 * ASTR, ASTR, lane, kc * 32);
      acc[s] = __builtin_amdgcn_wmma_f32_16x16x32_f16(false, a, false, b,
                                                      (short)0, acc[s], false, false);
    }
  }

  // ---- bias + ReLU -> f16 hidden tile in LDS ----
  const float bb1 = b1[n];
#pragma unroll
  for (int s = 0; s < NSUB; ++s) {
#pragma unroll
    for (int v = 0; v < 8; ++v) {
      float h0 = acc[s][v] + bb1; h0 = h0 > 0.0f ? h0 : 0.0f;
      hbuf[(s * 16 + mh + v) * HSTR + n] = (_Float16)h0;
    }
  }
  __syncthreads();

  // ---- GEMM2: h[64x128] @ W2[128x128] ----
  v8f dcc[NSUB];
#pragma unroll
  for (int s = 0; s < NSUB; ++s) dcc[s] = (v8f){};
#pragma unroll
  for (int kc = 0; kc < KC2; ++kc) {
    const v16h b = load_b_frag(pack2 + (((kc * 8 + w) * 32 + lane) << 4));
#pragma unroll
    for (int s = 0; s < NSUB; ++s) {
      const v16h a = load_a_frag(hbuf + s * 16 * HSTR, HSTR, lane, kc * 32);
      dcc[s] = __builtin_amdgcn_wmma_f32_16x16x32_f16(false, a, false, b,
                                                      (short)0, dcc[s], false, false);
    }
  }

  // ---- bias + store f32 output ----
  const float bb2 = b2[n];
#pragma unroll
  for (int s = 0; s < NSUB; ++s) {
#pragma unroll
    for (int v = 0; v < 8; ++v) {
      const int e0 = tile + s * 16 + mh + v;
      if (e0 < E) out[e0 * FD + n] = dcc[s][v] + bb2;
    }
  }
}

extern "C" void kernel_launch(void* const* d_in, const int* in_sizes, int n_in,
                              void* d_out, int out_size, void* d_ws, size_t ws_size,
                              hipStream_t stream) {
  const float* lig = (const float*)d_in[0];
  const float* pCA = (const float*)d_in[1];
  const float* pCb = (const float*)d_in[2];
  const float* pC  = (const float*)d_in[3];
  const float* pO  = (const float*)d_in[4];
  const float* pN  = (const float*)d_in[5];
  const float* W1  = (const float*)d_in[6];
  const float* b1  = (const float*)d_in[7];
  const float* W2  = (const float*)d_in[8];
  const float* b2  = (const float*)d_in[9];
  const int*   src = (const int*)d_in[10];
  const int*   dst = (const int*)d_in[11];
  const int E = in_sizes[10];

  _Float16* pack1 = (_Float16*)d_ws;           // 81920 B
  _Float16* pack2 = pack1 + PK1;               // 32768 B  (ws >= 114688 B)

  pack_weights<<<(PK1 + PK2 + 255) / 256, 256, 0, stream>>>(W1, W2, pack1, pack2);

  const int nblocks = (E + MT - 1) / MT;       // 7813 for E = 500000
  fused_edge_mlp<<<nblocks, 256, 0, stream>>>(lig, pCA, pCb, pC, pO, pN,
                                              b1, b2, src, dst,
                                              pack1, pack2, (float*)d_out, E);
}
